// Map_64974265254390
// MI455X (gfx1250) — matrix operand order repaired and checked by reading, not compiled, not run on hardware
//
#include <hip/hip_runtime.h>
#include <hip/hip_bf16.h>

// Instant-NGP fused inference for gfx1250 (MI455X), wave32 + WMMA f16.
//   - hash-grid encode (tables L2-resident: 64MB < 192MB L2)
//   - density MLP 32->64->16 (sigmoid)
//   - dir encoding (4 freq sin/cos)
//   - colour MLP 40(pad64)->64->64->3(pad16) (sigmoid)
// One wave == one 16-point tile; all GEMMs via v_wmma_f32_16x16x32_f16.

#define L_LEVELS 16
#define T_SIZE   524288      // 2^19, hash table entries per level
#define WAVES    4
#define BLOCK    (WAVES * 32)

typedef __attribute__((ext_vector_type(16))) _Float16 v16h;
typedef __attribute__((ext_vector_type(8)))  float    v8f;
typedef __attribute__((ext_vector_type(4)))  unsigned int v4u;

// floor(16 * 2^(l/3)) for l=0..15 (b = 32^(1/15) = 2^(1/3) exactly)
__constant__ float c_res[16] = {16.f, 20.f, 25.f, 32.f, 40.f, 50.f, 64.f, 80.f,
                                101.f, 128.f, 161.f, 203.f, 256.f, 322.f, 406.f, 512.f};

__device__ __forceinline__ void sched_fence() {
  // LDS ops are in-order within a wave; this only stops the compiler
  // from reordering cross-lane LDS producer/consumer pairs.
  __builtin_amdgcn_wave_barrier();
}

// A-matrix fragment (16x32 f16): lane holds row (lane&15),
// halfs [kb..kb+7] and [kb+16..kb+23], kb = kchunk*32 + (lane>>4)*8.
__device__ __forceinline__ v16h load_fragA(const _Float16* mat, int ldm, int lane, int kchunk) {
  const int row = lane & 15;
  const int kb  = kchunk * 32 + ((lane >> 4) << 3);
  const _Float16* p = mat + row * ldm + kb;
  union { v16h h; v4u u[2]; } u;
  u.u[0] = *(const v4u*)(p);
  u.u[1] = *(const v4u*)(p + 16);
  return u.h;
}

// B-matrix fragment (32x16 f16): lane holds column (lane&15) of tile ntile,
// K [kb..kb+7] and [kb+16..kb+23], kb = kchunk*32 + (lane>>4)*8. Weights stored (out,in)
// row-major == B^T, so each 8-half K run is contiguous in memory.
__device__ __forceinline__ v16h load_fragB(const _Float16* w, int ldm, int lane, int ntile, int kchunk) {
  const int row = ntile * 16 + (lane & 15);
  const int kb  = kchunk * 32 + ((lane >> 4) << 3);
  const _Float16* p = w + row * ldm + kb;
  union { v16h h; v4u u[2]; } u;
  u.u[0] = *(const v4u*)(p);
  u.u[1] = *(const v4u*)(p + 16);
  return u.h;
}

__device__ __forceinline__ v8f wmma32(v16h a, v16h b, v8f c) {
  return __builtin_amdgcn_wmma_f32_16x16x32_f16(false, a, false, b, (short)0, c, false, false);
}

// C/D layout: VGPR r -> row r + 8*(lane>>4), col = lane&15 (+16*ntile).
__device__ __forceinline__ void store_relu(_Float16* dst, int ldm, v8f c,
                                           const float* bias, int ntile, int lane) {
  const int col = ntile * 16 + (lane & 15);
  const float b = bias[col];
  const int mb = (lane >> 4) << 3;
#pragma unroll
  for (int r = 0; r < 8; ++r) {
    float v = c[r] + b;
    dst[(mb + r) * ldm + col] = (_Float16)(v > 0.f ? v : 0.f);
  }
}

// 64-in -> 64-out relu layer: 4 N-tiles x 2 K-chunks = 8 WMMAs.
__device__ __forceinline__ void layer64x64(const _Float16* actIn, const _Float16* w,
                                           const float* bias, _Float16* actOut, int lane) {
  v16h a0 = load_fragA(actIn, 64, lane, 0);
  v16h a1 = load_fragA(actIn, 64, lane, 1);
#pragma unroll
  for (int t = 0; t < 4; ++t) {
    v8f c = {};
    c = wmma32(a0, load_fragB(w, 64, lane, t, 0), c);
    c = wmma32(a1, load_fragB(w, 64, lane, t, 1), c);
    store_relu(actOut, 64, c, bias, t, lane);
  }
}

__global__ __launch_bounds__(BLOCK) void ngp_fused(
    const float* __restrict__ X, const float* __restrict__ D,
    const float* __restrict__ tables,
    const float* __restrict__ w1,  const float* __restrict__ b1,
    const float* __restrict__ w2,  const float* __restrict__ b2,
    const float* __restrict__ cw1, const float* __restrict__ cb1,
    const float* __restrict__ cw2, const float* __restrict__ cb2,
    const float* __restrict__ cw3, const float* __restrict__ cb3,
    float* __restrict__ out, int N, int R)
{
  __shared__ _Float16 s_w1[64 * 32];   // density L1 (64,32)
  __shared__ _Float16 s_w2[16 * 64];   // density L2 (16,64)
  __shared__ _Float16 s_cw1[64 * 64];  // colour L1 (64,40) padded K->64
  __shared__ _Float16 s_cw2[64 * 64];  // colour L2 (64,64)
  __shared__ _Float16 s_cw3[16 * 64];  // colour L3 (3,64) padded N->16
  __shared__ float s_b1[64], s_b2[16], s_cb1[64], s_cb2[64], s_cb3[16];
  __shared__ _Float16 s_act[WAVES][2][16 * 64];  // per-wave ping-pong

  const int tid = threadIdx.x;

  // ---- stage weights: f32 -> f16, with zero padding ----
  for (int i = tid; i < 64 * 32; i += BLOCK) s_w1[i] = (_Float16)w1[i];
  for (int i = tid; i < 16 * 64; i += BLOCK) s_w2[i] = (_Float16)w2[i];
  for (int i = tid; i < 64 * 64; i += BLOCK) {
    const int nn = i >> 6, kk = i & 63;
    s_cw1[i] = (kk < 40) ? (_Float16)cw1[nn * 40 + kk] : (_Float16)0.f;
    s_cw2[i] = (_Float16)cw2[i];
  }
  for (int i = tid; i < 16 * 64; i += BLOCK) {
    const int nn = i >> 6, kk = i & 63;
    s_cw3[i] = (nn < 3) ? (_Float16)cw3[nn * 64 + kk] : (_Float16)0.f;
  }
  if (tid < 64) { s_b1[tid] = b1[tid]; s_cb1[tid] = cb1[tid]; s_cb2[tid] = cb2[tid]; }
  if (tid < 16) { s_b2[tid] = b2[tid]; s_cb3[tid] = (tid < 3) ? cb3[tid] : 0.f; }
  __syncthreads();

  const int lane = tid & 31;
  const int wave = tid >> 5;
  const int tileBase = (blockIdx.x * WAVES + wave) * 16;
  if (tileBase >= N) return;

  _Float16* bufA = s_act[wave][0];
  _Float16* bufB = s_act[wave][1];

  // ---- hash-grid encoding -> bufA as 16x32 f16 (row-major, ldm 32) ----
  // lanes 0-15 do levels 0-7 of point (lane&15); lanes 16-31 do levels 8-15.
  const int p = lane & 15;
  const int n = tileBase + p;
  const float x0 = X[3 * n + 0], x1 = X[3 * n + 1], x2 = X[3 * n + 2];
  const int lb = (lane >> 4) << 3;
#pragma unroll
  for (int li = 0; li < 8; ++li) {
    const int lvl = lb + li;
    const float res = c_res[lvl];
    const float s0 = x0 * res, s1 = x1 * res, s2 = x2 * res;
    const float p0 = floorf(s0), p1 = floorf(s1), p2 = floorf(s2);
    const float r0 = s0 - p0, r1 = s1 - p1, r2 = s2 - p2;
    const unsigned int i0 = (unsigned int)(int)p0;
    const unsigned int i1 = (unsigned int)(int)p1;
    const unsigned int i2 = (unsigned int)(int)p2;
    const float2* tbl = (const float2*)tables + (size_t)lvl * T_SIZE;
    float a0 = 0.f, a1 = 0.f;
#pragma unroll
    for (int cn = 0; cn < 8; ++cn) {
      const unsigned int u0 = i0 + ((cn >> 2) & 1);
      const unsigned int u1 = i1 + ((cn >> 1) & 1);
      const unsigned int u2 = i2 + (cn & 1);
      const unsigned int idx = (u0 ^ (u1 * 2654435761u) ^ (u2 * 805459861u)) & (T_SIZE - 1);
      const float2 f = tbl[idx];                 // 8B gather, L2-resident
      const float w = ((cn & 4) ? r0 : 1.f - r0)
                    * ((cn & 2) ? r1 : 1.f - r1)
                    * ((cn & 1) ? r2 : 1.f - r2);
      a0 += f.x * w; a1 += f.y * w;
    }
    union { _Float16 h[2]; unsigned int u; } pk;
    pk.h[0] = (_Float16)a0; pk.h[1] = (_Float16)a1;
    *(unsigned int*)(bufA + p * 32 + 2 * lvl) = pk.u;
  }
  sched_fence();

  // ---- density layer 1: 32 -> 64, relu (4 WMMAs), bufA -> bufB ----
  {
    v16h a = load_fragA(bufA, 32, lane, 0);
#pragma unroll
    for (int t = 0; t < 4; ++t) {
      v8f c = {};
      c = wmma32(a, load_fragB(s_w1, 32, lane, t, 0), c);
      store_relu(bufB, 64, c, s_b1, t, lane);
    }
  }
  sched_fence();

  // ---- direction encoding + zero pad into bufA (now 16x64 c-matrix) ----
  // cols 0-15: density (written below); 16-39: sin/cos; 40-63: zeros.
  {
    const int rIdx = n % R;  // tiled direction: row n uses D[n % R]
    const float d0 = D[3 * rIdx + 0], d1 = D[3 * rIdx + 1], d2 = D[3 * rIdx + 2];
    const float PI = 3.14159265358979323846f;
    const int g = lane >> 4;
#pragma unroll
    for (int fi = 0; fi < 2; ++fi) {
      const int f = g * 2 + fi;
      const float s = (float)(1 << f) * PI;
      const int base = p * 64 + 16 + f * 6;
      bufA[base + 0] = (_Float16)__sinf(s * d0);
      bufA[base + 1] = (_Float16)__sinf(s * d1);
      bufA[base + 2] = (_Float16)__sinf(s * d2);
      bufA[base + 3] = (_Float16)__cosf(s * d0);
      bufA[base + 4] = (_Float16)__cosf(s * d1);
      bufA[base + 5] = (_Float16)__cosf(s * d2);
    }
    const int zb = p * 64 + 40 + g * 12;
#pragma unroll
    for (int z = 0; z < 12; ++z) bufA[zb + z] = (_Float16)0.f;
  }

  // ---- density layer 2: 64 -> 16, sigmoid (2 WMMAs) ----
  // writes sigmoid to bufA cols 0-15 and density[:,0] to out[0..N).
  {
    v16h a0 = load_fragA(bufB, 64, lane, 0);
    v16h a1 = load_fragA(bufB, 64, lane, 1);
    v8f c = {};
    c = wmma32(a0, load_fragB(s_w2, 64, lane, 0, 0), c);
    c = wmma32(a1, load_fragB(s_w2, 64, lane, 0, 1), c);
    const int col = lane & 15;
    const float bb = s_b2[col];
    const int mb = (lane >> 4) << 3;
#pragma unroll
    for (int r = 0; r < 8; ++r) {
      const float s = 1.f / (1.f + __expf(-(c[r] + bb)));
      bufA[(mb + r) * 64 + col] = (_Float16)s;
      if (col == 0) out[tileBase + mb + r] = s;
    }
  }
  sched_fence();

  // ---- colour layer 1: 64(pad from 40) -> 64 relu, bufA -> bufB (8 WMMAs) ----
  layer64x64(bufA, s_cw1, s_cb1, bufB, lane);
  sched_fence();

  // ---- colour layer 2: 64 -> 64 relu, bufB -> bufA (8 WMMAs) ----
  layer64x64(bufB, s_cw2, s_cb2, bufA, lane);
  sched_fence();

  // ---- colour layer 3: 64 -> 16(3 valid), sigmoid (2 WMMAs) ----
  {
    v16h a0 = load_fragA(bufA, 64, lane, 0);
    v16h a1 = load_fragA(bufA, 64, lane, 1);
    v8f c = {};
    c = wmma32(a0, load_fragB(s_cw3, 64, lane, 0, 0), c);
    c = wmma32(a1, load_fragB(s_cw3, 64, lane, 0, 1), c);
    const int col = lane & 15;
    if (col < 3) {
      const float bb = s_cb3[col];
      const int mb = (lane >> 4) << 3;
#pragma unroll
      for (int r = 0; r < 8; ++r) {
        const float s = 1.f / (1.f + __expf(-(c[r] + bb)));
        out[(size_t)N + (size_t)(tileBase + mb + r) * 3 + col] = s;
      }
    }
  }
}

extern "C" void kernel_launch(void* const* d_in, const int* in_sizes, int n_in,
                              void* d_out, int out_size, void* d_ws, size_t ws_size,
                              hipStream_t stream) {
  // inputs: X, D, M, tables, w1, b1, w2, b2, cw1, cb1, cw2, cb2, cw3, cb3
  const float* X   = (const float*)d_in[0];
  const float* D   = (const float*)d_in[1];
  const float* tab = (const float*)d_in[3];
  const float* w1  = (const float*)d_in[4];
  const float* b1  = (const float*)d_in[5];
  const float* w2  = (const float*)d_in[6];
  const float* b2  = (const float*)d_in[7];
  const float* cw1 = (const float*)d_in[8];
  const float* cb1 = (const float*)d_in[9];
  const float* cw2 = (const float*)d_in[10];
  const float* cb2 = (const float*)d_in[11];
  const float* cw3 = (const float*)d_in[12];
  const float* cb3 = (const float*)d_in[13];

  const int N = in_sizes[0] / 3;   // 1,048,576 points
  const int R = in_sizes[1] / 3;   // 32,768 rays

  const int pointsPerBlock = WAVES * 16;
  const int blocks = (N + pointsPerBlock - 1) / pointsPerBlock;
  ngp_fused<<<blocks, BLOCK, 0, stream>>>(X, D, tab, w1, b1, w2, b2,
                                          cw1, cb1, cw2, cb2, cw3, cb3,
                                          (float*)d_out, N, R);
}
